// Controller_86655260164956
// MI455X (gfx1250) — matrix operand order repaired and checked by reading, not compiled, not run on hardware
//
#include <hip/hip_runtime.h>
#include <hip/hip_bf16.h>

typedef __attribute__((ext_vector_type(16))) _Float16 v16h;
typedef __attribute__((ext_vector_type(8)))  _Float16 v8h;
typedef __attribute__((ext_vector_type(8)))  float    v8f;
typedef __attribute__((ext_vector_type(4)))  unsigned int u32x4;
typedef __attribute__((ext_vector_type(8)))  int       i32x8;
typedef __attribute__((ext_vector_type(4)))  int       i32x4;

#define Hdim   128
#define FOURH  512
#define NBR    8
#define LSTEPS 64
#define BT     16      // batch rows per workgroup

// ---- LDS layout (bytes) ----
#define OFF_WIH    0          // [512][128] f16  = 131072
#define OFF_WHH    131072     // [512][128] f16  = 131072  (contiguous with WIH)
#define OFF_GATES  262144     // [16][512]  f32  = 32768
#define OFF_XS     294912     // [16][128]  f16  = 4096
#define OFF_HS     299008     // [16][128]  f16  = 4096
#define OFF_X0     303104     // [16][128]  f16  = 4096
#define OFF_WSOFT  307200     // [8][128]   f32  = 4096
#define OFF_WEMB   311296     // [8][128]   f32  = 4096
#define OFF_BIAS   315392     // [512]      f32  = 2048
#define OFF_LOGIT  317440     // [16][8]    f32  = 512
#define OFF_SCORE  317952     // [16][8]    f32  = 512
#define OFF_BR     318464     // [16]       i32  = 64
#define SMEM_BYTES 318528

__device__ __forceinline__ float sigmoidf_(float x) { return 1.0f / (1.0f + __expf(-x)); }

// Prep kernel: fp32 weights -> f16 workspace (W_ih then W_hh, contiguous [1024,128])
__global__ __launch_bounds__(256)
void cvt_weights_f16(const float* __restrict__ wih, const float* __restrict__ whh,
                     _Float16* __restrict__ dst) {
    int i = blockIdx.x * 256 + threadIdx.x;           // 0 .. 131071
    if (i < FOURH * Hdim)
        dst[i] = (_Float16)wih[i];
    else
        dst[i] = (_Float16)whh[i - FOURH * Hdim];
}

// TDM weight load: one DMA of the whole [1024,128] f16 block into LDS offset 0.
// D# built per CDNA5 ISA (2D tensor, tile == tensor).
__device__ __forceinline__ void tdm_load_weights(const _Float16* src) {
#if __has_builtin(__builtin_amdgcn_tensor_load_to_lds)
    const unsigned long long ga = (unsigned long long)src;
    u32x4 g0;
    g0[0] = 1u;                                        // count=1, is_restore=0
    g0[1] = (unsigned)OFF_WIH;                         // lds_addr (bytes)
    g0[2] = (unsigned)(ga & 0xFFFFFFFFull);            // global_addr[31:0]
    g0[3] = (unsigned)((ga >> 32) & 0x1FFFFFFull)      // global_addr[56:32]
          | (2u << 30);                                // type = 2 ("image")
    i32x8 g1;
    g1[0] = (int)(1u << 16);                           // wg_mask=0, data_size=1 (2B)
    g1[1] = (int)(128u << 16);                         // tensor_dim0[15:0]=128
    g1[2] = (int)(1024u << 16);                        // dim0 hi=0 | tensor_dim1[15:0]=1024
    g1[3] = (int)(128u << 16);                         // dim1 hi=0 | tile_dim0=128
    g1[4] = (int)1024u;                                // tile_dim1=1024 | tile_dim2=0
    g1[5] = (int)128u;                                 // tensor_dim0_stride=128
    g1[6] = 0;                                         // stride hi / dim1_stride lo
    g1[7] = 0;
    i32x4 z4 = {0, 0, 0, 0};
#if __clang_major__ >= 23
    i32x8 z8 = {0, 0, 0, 0, 0, 0, 0, 0};
    __builtin_amdgcn_tensor_load_to_lds(g0, g1, z4, z4, z8, 0);
#else
    __builtin_amdgcn_tensor_load_to_lds(g0, g1, z4, z4, 0);
#endif
    __builtin_amdgcn_s_wait_tensorcnt(0);
#else
    (void)src;
#endif
}

__global__ __launch_bounds__(256)
void lstm_controller_wmma(const int*   __restrict__ class_ids,
                          const float* __restrict__ gumbel_u,   // [L,B,8]
                          const float* __restrict__ g_emb,      // [NC,128]
                          const float* __restrict__ w_emb,      // [8,128]
                          const float* __restrict__ w_soft,     // [8,128]
                          const float* __restrict__ b_ih,       // [512]
                          const float* __restrict__ b_hh,       // [512]
                          const _Float16* __restrict__ wih_h,   // [1024,128] f16 (ih|hh)
                          float* __restrict__ out, int B)
{
    extern __shared__ char smem[];
    _Float16* wih   = (_Float16*)(smem + OFF_WIH);
    _Float16* whh   = (_Float16*)(smem + OFF_WHH);
    float*    gates = (float*)   (smem + OFF_GATES);
    _Float16* xs    = (_Float16*)(smem + OFF_XS);
    _Float16* hs    = (_Float16*)(smem + OFF_HS);
    _Float16* x0s   = (_Float16*)(smem + OFF_X0);
    float*    wsoft = (float*)   (smem + OFF_WSOFT);
    float*    wembs = (float*)   (smem + OFF_WEMB);
    float*    bias  = (float*)   (smem + OFF_BIAS);
    float*    logit_s = (float*) (smem + OFF_LOGIT);
    float*    score_s = (float*) (smem + OFF_SCORE);
    int*      branch_s = (int*)  (smem + OFF_BR);

    const int tid  = threadIdx.x;
    const int lane = tid & 31;
    const int wv   = tid >> 5;                 // wave 0..7
    const int b0   = blockIdx.x * BT;

    // ---- one-time: 256KB weights global->LDS ----
#if __has_builtin(__builtin_amdgcn_tensor_load_to_lds)
    if (wv == 0) tdm_load_weights(wih_h);      // single TDM DMA from wave 0
#else
    {   // cooperative fallback, vectorized 16B
        const uint4* s0 = (const uint4*)wih_h;  uint4* d0 = (uint4*)wih;
        #pragma unroll 4
        for (int i = tid; i < (2 * FOURH * Hdim) / 8; i += 256) d0[i] = s0[i];
    }
#endif
    for (int i = tid; i < FOURH; i += 256) bias[i] = b_ih[i] + b_hh[i];
    for (int i = tid; i < NBR * Hdim; i += 256) { wsoft[i] = w_soft[i]; wembs[i] = w_emb[i]; }

    // x0 = g_emb[class_ids];  x = x0;  h = 0
    for (int i = tid; i < BT * Hdim; i += 256) {
        int row = i >> 7, unit = i & 127;
        int cid = class_ids[b0 + row];
        _Float16 v = (_Float16)g_emb[(long long)cid * Hdim + unit];
        x0s[i] = v; xs[i] = v; hs[i] = (_Float16)0.0f;
    }

    float cst[8];                               // cell state: registers only, all 64 steps
    #pragma unroll
    for (int p = 0; p < 8; ++p) cst[p] = 0.0f;

    const int m  = lane & 15;                   // A-matrix row for this lane
    const int hi = lane >> 4;                   // lane half selects K sub-range
    const long long BL = (long long)B * LSTEPS;

    for (int t = 0; t < LSTEPS; ++t) {
        __syncthreads();                        // x/h tiles (and, first pass, weights) ready

        // ---- A fragments (x and h), hoisted once per step per wave ----
        // A 16x32 in this code's contiguous K order (B matches): lanes hold 16 contiguous K {kb..kb+15}, kb = 32*kc + 16*hi
        v16h ax[4], ah[4];
        #pragma unroll
        for (int kc = 0; kc < 4; ++kc) {
            const int kb = kc * 32 + hi * 16;
            union { v16h v; v8h h2[2]; } ua, uh;
            ua.h2[0] = *(const v8h*)(xs + m * Hdim + kb);
            ua.h2[1] = *(const v8h*)(xs + m * Hdim + kb + 8);
            uh.h2[0] = *(const v8h*)(hs + m * Hdim + kb);
            uh.h2[1] = *(const v8h*)(hs + m * Hdim + kb + 8);
            ax[kc] = ua.v; ah[kc] = uh.v;
        }

        // ---- gates tile [16 x 512]: wave wv owns columns [64*wv, 64*wv+64) ----
        // 8 independent accumulator chains (4 tiles x {x-part, h-part}) for WMMA ILP;
        // kc outer / tile inner so consecutive WMMAs are independent.
        v8f accx[4], acch[4];
        #pragma unroll
        for (int j = 0; j < 4; ++j) { v8f z = {}; accx[j] = z; acch[j] = z; }

        const int kr = hi * 16;                 // B: 16 contiguous K per lane
        #pragma unroll
        for (int kc = 0; kc < 4; ++kc) {
            #pragma unroll
            for (int j = 0; j < 4; ++j) {
                const int n = (wv * 4 + j) * 16 + (lane & 15);
                v16h bih = *(const v16h*)(wih + n * Hdim + kc * 32 + kr);
                v16h bhh = *(const v16h*)(whh + n * Hdim + kc * 32 + kr);
                accx[j] = __builtin_amdgcn_wmma_f32_16x16x32_f16(false, ax[kc], false, bih,
                                                                 (short)0, accx[j], false, false);
                acch[j] = __builtin_amdgcn_wmma_f32_16x16x32_f16(false, ah[kc], false, bhh,
                                                                 (short)0, acch[j], false, false);
            }
        }
        #pragma unroll
        for (int j = 0; j < 4; ++j) {
            const v8f acc = accx[j] + acch[j];
            const int n = (wv * 4 + j) * 16 + (lane & 15);
            const int mb = hi * 8;              // C layout: VGPR r -> M = r + 8*hi, N = lane&15
            #pragma unroll
            for (int r = 0; r < 8; ++r)
                gates[(mb + r) * FOURH + n] = acc[r];
        }
        __syncthreads();

        // ---- LSTM elementwise (gate order i,f,g,o); c in registers ----
        #pragma unroll
        for (int p = 0; p < 8; ++p) {
            const int idx = tid + 256 * p;
            const int row = idx >> 7, unit = idx & 127;
            const float gi = gates[row * FOURH +       unit] + bias[      unit];
            const float gf = gates[row * FOURH + 128 + unit] + bias[128 + unit];
            const float gg = gates[row * FOURH + 256 + unit] + bias[256 + unit];
            const float go = gates[row * FOURH + 384 + unit] + bias[384 + unit];
            const float c2 = sigmoidf_(gf) * cst[p] + sigmoidf_(gi) * tanhf(gg);
            cst[p] = c2;
            hs[idx] = (_Float16)(sigmoidf_(go) * tanhf(c2));
        }
        __syncthreads();

        // ---- softmax head: logit = 2.5*tanh((h2 @ w_softT)/5) + gumbel ----
        if (tid < BT * NBR) {
            const int row = tid >> 3, nb = tid & 7;
            float s = 0.0f;
            #pragma unroll 8
            for (int k = 0; k < Hdim; ++k)
                s += (float)hs[row * Hdim + k] * wsoft[nb * Hdim + k];
            const float lg = 2.5f * tanhf(s * 0.2f);
            float u = gumbel_u[((long long)t * B + (b0 + row)) * NBR + nb];
            u = fminf(fmaxf(u, 1e-8f), 1.0f - 1e-8f);
            logit_s[tid] = lg;
            score_s[tid] = lg - __logf(-__logf(u));
        }
        if (t + 1 < LSTEPS && tid < BT)
            __builtin_prefetch(&gumbel_u[((long long)(t + 1) * B + (b0 + tid)) * NBR], 0, 1);
        __syncthreads();

        // ---- argmax / log_softmax / entropy / outputs ----
        if (tid < BT) {
            const int row = tid;
            int br = 0;
            float best = score_s[row * NBR], mx = logit_s[row * NBR];
            #pragma unroll
            for (int nb = 1; nb < NBR; ++nb) {
                const float sc = score_s[row * NBR + nb];
                if (sc > best) { best = sc; br = nb; }
                mx = fmaxf(mx, logit_s[row * NBR + nb]);
            }
            float se = 0.0f;
            #pragma unroll
            for (int nb = 0; nb < NBR; ++nb) se += __expf(logit_s[row * NBR + nb] - mx);
            const float lse = mx + __logf(se);
            const float lp  = logit_s[row * NBR + br] - lse;
            float ent = 0.0f;
            #pragma unroll
            for (int nb = 0; nb < NBR; ++nb) {
                const float l = logit_s[row * NBR + nb] - lse;
                ent -= l * __expf(l);
            }
            const long long o = (long long)(b0 + row) * LSTEPS + t;
            out[o]           = (float)br;     // sample_arc (as float)
            out[BL + o]      = lp;            // sample_log_prob
            out[2 * BL + o]  = ent;           // sample_entropy
            out[3 * BL + o]  = __expf(lp);    // sample_prob
            branch_s[row] = br;
        }
        __syncthreads();

        // ---- x = (w_emb[branch] + x0) * 0.5 ----
        #pragma unroll
        for (int p = 0; p < 8; ++p) {
            const int idx = tid + 256 * p;
            const int row = idx >> 7, unit = idx & 127;
            const int br = branch_s[row];
            xs[idx] = (_Float16)((wembs[br * Hdim + unit] + (float)x0s[idx]) * 0.5f);
        }
    }
}

extern "C" void kernel_launch(void* const* d_in, const int* in_sizes, int n_in,
                              void* d_out, int out_size, void* d_ws, size_t ws_size,
                              hipStream_t stream) {
    const int*   class_ids = (const int*)  d_in[0];
    const float* gumbel_u  = (const float*)d_in[1];
    const float* g_emb     = (const float*)d_in[2];
    const float* w_emb     = (const float*)d_in[3];
    const float* w_soft    = (const float*)d_in[4];
    const float* w_ih      = (const float*)d_in[5];
    const float* w_hh      = (const float*)d_in[6];
    const float* b_ih      = (const float*)d_in[7];
    const float* b_hh      = (const float*)d_in[8];
    float* out = (float*)d_out;
    const int B = in_sizes[0];                       // 16384

    _Float16* w16 = (_Float16*)d_ws;                 // 256 KB: [W_ih f16 | W_hh f16]
    cvt_weights_f16<<<(2 * FOURH * Hdim) / 256, 256, 0, stream>>>(w_ih, w_hh, w16);

    lstm_controller_wmma<<<B / BT, 256, SMEM_BYTES, stream>>>(
        class_ids, gumbel_u, g_emb, w_emb, w_soft, b_ih, b_hh, w16, out, B);
}
